// AutoEncoder_46067819217111
// MI455X (gfx1250) — compile-verified, matrix operand order checked by reading, not run on hardware
//
#include <hip/hip_runtime.h>
#include <math.h>

typedef __attribute__((ext_vector_type(16))) __bf16 v16bf;
typedef __attribute__((ext_vector_type(8)))  float  v8f;
typedef __attribute__((ext_vector_type(4)))  unsigned int v4u;
typedef __attribute__((ext_vector_type(8)))  int v8i;
typedef __attribute__((ext_vector_type(4)))  int v4i;

#if defined(__has_builtin)
#  if __has_builtin(__builtin_amdgcn_tensor_load_to_lds)
#    define USE_TDM 1
#  endif
#endif
#ifndef USE_TDM
#  define USE_TDM 0
#endif

#define DEVINL static __device__ __forceinline__

DEVINL unsigned short f2us_bf16(float f) {
  unsigned int u = __float_as_uint(f);
  u += 0x7FFFu + ((u >> 16) & 1u);   // round-to-nearest-even
  return (unsigned short)(u >> 16);
}
DEVINL __bf16 us2bf(unsigned short s) {
  union { unsigned short u; __bf16 b; } c; c.u = s; return c.b;
}
union FragU { v16bf v; unsigned int u[8]; };

// ---------------------------------------------------------------------------
// Layout / elementwise helpers
// ---------------------------------------------------------------------------
__global__ void k_nchw_to_nhwc_bf16(const float* __restrict__ in,
                                    unsigned short* __restrict__ out,
                                    int N, int C, int H, int W) {
  long long i = (long long)blockIdx.x * blockDim.x + threadIdx.x;
  long long tot = (long long)N * C * H * W;
  if (i >= tot) return;
  int c = (int)(i % C); long long r = i / C;
  int w = (int)(r % W); r /= W;
  int h = (int)(r % H); int n = (int)(r / H);
  out[i] = f2us_bf16(in[(((long long)n * C + c) * H + h) * W + w]);
}

__global__ void k_nhwc_to_nchw(const float* __restrict__ in,
                               float* __restrict__ out,
                               int N, int C, int H, int W) {
  long long i = (long long)blockIdx.x * blockDim.x + threadIdx.x;
  long long tot = (long long)N * C * H * W;
  if (i >= tot) return;
  int w = (int)(i % W); long long r = i / W;
  int h = (int)(r % H); r /= H;
  int c = (int)(r % C); int n = (int)(r / C);
  out[i] = in[(((long long)n * H + h) * W + w) * C + c];
}

__global__ void k_convert(const float* __restrict__ in,
                          float* __restrict__ of32,
                          unsigned short* __restrict__ obf,
                          long long tot, int relu) {
  long long i = (long long)blockIdx.x * blockDim.x + threadIdx.x;
  if (i >= tot) return;
  float v = in[i];
  if (relu) v = fmaxf(v, 0.f);
  if (of32) of32[i] = v;
  if (obf)  obf[i] = f2us_bf16(v);
}

// ---------------------------------------------------------------------------
// BatchNorm (training mode statistics over rows of NHWC)
// ---------------------------------------------------------------------------
__global__ void k_bn_stats(const float* __restrict__ x,
                           float* __restrict__ stats, int M, int C) {
  int c = threadIdx.x;                       // blockDim.x == C == 256
  int rows = (M + gridDim.x - 1) / gridDim.x;
  int r0 = blockIdx.x * rows;
  int r1 = r0 + rows; if (r1 > M) r1 = M;
  float s = 0.f, sq = 0.f;
  for (int r = r0; r < r1; ++r) {
    float v = x[(long long)r * C + c];
    s += v; sq += v * v;
  }
  atomicAdd(&stats[c], s);
  atomicAdd(&stats[C + c], sq);
}

__global__ void k_bn_apply(const float* __restrict__ x,
                           const float* __restrict__ stats,
                           const float* __restrict__ g,
                           const float* __restrict__ b,
                           float* __restrict__ of32,
                           unsigned short* __restrict__ obf,
                           long long tot, int C, float inv_m, int relu) {
  long long i = (long long)blockIdx.x * blockDim.x + threadIdx.x;
  if (i >= tot) return;
  int c = (int)(i % C);
  float mean = stats[c] * inv_m;
  float var  = stats[C + c] * inv_m - mean * mean;
  float v = (x[i] - mean) * rsqrtf(var + 1e-5f) * g[c] + b[c];
  if (relu) v = fmaxf(v, 0.f);
  if (of32) of32[i] = v;
  if (obf)  obf[i] = f2us_bf16(v);
}

// ---------------------------------------------------------------------------
// Weight packing
// ---------------------------------------------------------------------------
__global__ void k_prep_wconv(const float* __restrict__ w,
                             unsigned short* __restrict__ Bg,
                             int Cout, int Cin, int KH, int KW, int Kpad) {
  int i = blockIdx.x * blockDim.x + threadIdx.x;
  int tot = Kpad * Cout;
  if (i >= tot) return;
  int o = i % Cout, k = i / Cout;
  int Ktot = Cin * KH * KW;
  float v = 0.f;
  if (k < Ktot) {
    int ci = k % Cin; int t = k / Cin;
    int kw = t % KW;  int kh = t / KW;
    v = w[(((long long)o * Cin + ci) * KH + kh) * KW + kw];
  }
  Bg[(long long)k * Cout + o] = f2us_bf16(v);
}

// ConvTranspose (Cin,Cout,4,4), stride 2, pad 1: 4 phase-specific 2x2 kernels.
__global__ void k_prep_wdeconv(const float* __restrict__ w,
                               unsigned short* __restrict__ Bg,
                               int Cin, int Cout) {
  int ph = blockIdx.y;
  int py = ph >> 1, px = ph & 1;
  int i = blockIdx.x * blockDim.x + threadIdx.x;
  int Kpad = 4 * Cin;
  int tot = Kpad * Cout;
  if (i >= tot) return;
  int o = i % Cout, k = i / Cout;
  int ci = k % Cin; int t = k / Cin;
  int tw = t & 1, th = t >> 1;
  int kh = (py == 0) ? (th ? 3 : 1) : (th ? 2 : 0);
  int kw = (px == 0) ? (tw ? 3 : 1) : (tw ? 2 : 0);
  float v = w[(((long long)ci * Cout + o) * 4 + kh) * 4 + kw];
  Bg[((long long)ph * Kpad + k) * Cout + o] = f2us_bf16(v);
}

// ---------------------------------------------------------------------------
// WMMA compute core: double-buffered LDS tiles
//   lA: [64 rows (M)][34] bf16, lB: [64 cols (N)][34] bf16 (transposed)
// ---------------------------------------------------------------------------
DEVINL void wmma_step(const unsigned short (*lA)[34],
                      const unsigned short (*lB)[34],
                      int wv, int grp, int lr, v8f acc[4]) {
  FragU af;
  const int mrow = wv * 16 + lr;
#pragma unroll
  for (int i = 0; i < 8; ++i) {
    int kk = 2 * i + ((i >= 4) ? 8 : 0) + grp * 8;
    af.u[i] = *(const unsigned int*)&lA[mrow][kk];
  }
#pragma unroll
  for (int j = 0; j < 4; ++j) {
    FragU bf;
    const int nrow = j * 16 + lr;
#pragma unroll
    for (int i = 0; i < 8; ++i) {
      int kk = 2 * i + ((i >= 4) ? 8 : 0) + grp * 8;
      bf.u[i] = *(const unsigned int*)&lB[nrow][kk];
    }
    acc[j] = __builtin_amdgcn_wmma_f32_16x16x32_bf16(
        false, af.v, false, bf.v, (short)0, acc[j], false, false);
  }
}

// ---------------------------------------------------------------------------
// Implicit-GEMM forward conv (templated shape, double-buffered, WMMA bf16)
// ---------------------------------------------------------------------------
template <int N, int H, int W, int Cin, int Cout, int KH, int KW, int ST, int PD>
__global__ __launch_bounds__(128)
void k_conv_wmma(const unsigned short* __restrict__ Xbf,  // NHWC bf16
                 const unsigned short* __restrict__ Bg,   // [Kpad][Cout] bf16
                 const float* __restrict__ bias,
                 const float* __restrict__ residual,      // NHWC fp32 or null
                 float* __restrict__ Out,                 // NHWC fp32
                 int relu) {
  constexpr int OH = (H + 2 * PD - KH) / ST + 1;
  constexpr int OW = (W + 2 * PD - KW) / ST + 1;
  constexpr int Mtot = N * OH * OW;
  constexpr int Ktot = Cin * KH * KW;
  constexpr int Kpad = (Ktot + 31) & ~31;
  constexpr int NSTEP = Kpad / 32;

  __shared__ unsigned short lA[2][64][34];
  __shared__ unsigned short lB[2][64][34];   // transposed: [n][k]
  const int tid = threadIdx.x;
  const int wv = tid >> 5, lane = tid & 31;
  const int grp = lane >> 4, lr = lane & 15;
  const int m0 = blockIdx.x * 64;
  const int n0 = blockIdx.y * 64;
  const int kkA = tid & 31;     // fixed k-lane for A loads
  const int rA0 = tid >> 5;     // starting row (step 4)
  const int cB  = tid & 63;     // fixed n-col for B loads
  const int krB0 = tid >> 6;    // starting k-row (step 2)

  v8f acc[4];
#pragma unroll
  for (int j = 0; j < 4; ++j)
#pragma unroll
    for (int e = 0; e < 8; ++e) acc[j][e] = 0.f;

  auto loadTiles = [&](int k0, int buf) {
    // ---- A tile: im2col gather; k-decode is loop invariant per thread ----
    const int k = k0 + kkA;
    const bool kv = (k < Ktot);
    const int ci = k % Cin;
    const int tt = k / Cin;
    const int kw = tt % KW;
    const int kh = tt / KW;
#pragma unroll
    for (int it = 0; it < 16; ++it) {
      int r = rA0 + it * 4;
      int m = m0 + r;
      unsigned short us = 0;
      if (kv && m < Mtot) {
        int ox = m % OW; int tm = m / OW;
        int oy = tm % OH; int n = tm / OH;
        int ix = ox * ST - PD + kw;
        int iy = oy * ST - PD + kh;
        if (ix >= 0 && ix < W && iy >= 0 && iy < H)
          us = Xbf[(((long long)n * H + iy) * W + ix) * Cin + ci];
      }
      lA[buf][r][kkA] = us;
    }
    // ---- B tile (stored transposed: [col][k]) ----
    const int cc = n0 + cB;
#pragma unroll
    for (int it = 0; it < 16; ++it) {
      int kr = krB0 + it * 2;
      unsigned short us = 0;
      if (cc < Cout) us = Bg[(long long)(k0 + kr) * Cout + cc];
      lB[buf][cB][kr] = us;
    }
  };

  loadTiles(0, 0);
  __syncthreads();
  for (int s = 0; s < NSTEP; ++s) {
    const int cur = s & 1;
    if (s + 1 < NSTEP) loadTiles((s + 1) * 32, cur ^ 1);
    wmma_step(lA[cur], lB[cur], wv, grp, lr, acc);
    __syncthreads();
  }

  // epilogue: bias + residual + relu, NHWC store
#pragma unroll
  for (int j = 0; j < 4; ++j) {
    int c = n0 + j * 16 + lr;
    if (c >= Cout) continue;
    float bb = bias ? bias[c] : 0.f;
#pragma unroll
    for (int v = 0; v < 8; ++v) {
      int m = m0 + wv * 16 + grp * 8 + v;
      if (m >= Mtot) continue;
      float val = acc[j][v] + bb;
      if (residual) val += residual[(long long)m * Cout + c];
      if (relu) val = fmaxf(val, 0.f);
      Out[(long long)m * Cout + c] = val;
    }
  }
}

// ---------------------------------------------------------------------------
// Phase-decomposed ConvTranspose2d (k=4, s=2, p=1): blockIdx.z = phase.
// ---------------------------------------------------------------------------
template <int N, int Hin, int Win, int Cin, int Cout>
__global__ __launch_bounds__(128)
void k_deconv_wmma(const unsigned short* __restrict__ Xbf,  // NHWC bf16
                   const unsigned short* __restrict__ Bg,   // 4 x [4*Cin][Cout]
                   const float* __restrict__ bias,
                   float* __restrict__ OutNHWC,
                   float* __restrict__ OutNCHW,             // with tanh
                   int tanh_flag) {
  constexpr int Mtot = N * Hin * Win;
  constexpr int Kpad = 4 * Cin;
  constexpr int NSTEP = Kpad / 32;

  __shared__ unsigned short lA[2][64][34];
  __shared__ unsigned short lB[2][64][34];
  const int tid = threadIdx.x;
  const int wv = tid >> 5, lane = tid & 31;
  const int grp = lane >> 4, lr = lane & 15;
  const int ph = blockIdx.z;
  const int py = ph >> 1, px = ph & 1;
  const int m0 = blockIdx.x * 64;
  const int n0 = blockIdx.y * 64;
  const unsigned short* Bp = Bg + (long long)ph * Kpad * Cout;
  const int kkA = tid & 31;
  const int rA0 = tid >> 5;
  const int cB  = tid & 63;
  const int krB0 = tid >> 6;

  v8f acc[4];
#pragma unroll
  for (int j = 0; j < 4; ++j)
#pragma unroll
    for (int e = 0; e < 8; ++e) acc[j][e] = 0.f;

  auto loadTiles = [&](int k0, int buf) {
    const int k = k0 + kkA;
    const int ci = k % Cin;
    const int tt = k / Cin;
    const int tw = tt & 1, th = tt >> 1;
    const int dy = (py == 0) ? (th ? -1 : 0) : (th ? 0 : 1);
    const int dx = (px == 0) ? (tw ? -1 : 0) : (tw ? 0 : 1);
#pragma unroll
    for (int it = 0; it < 16; ++it) {
      int r = rA0 + it * 4;
      int m = m0 + r;
      unsigned short us = 0;
      if (m < Mtot) {
        int x = m % Win; int tm = m / Win;
        int y = tm % Hin; int n = tm / Hin;
        int iy = y + dy, ix = x + dx;
        if (iy >= 0 && iy < Hin && ix >= 0 && ix < Win)
          us = Xbf[(((long long)n * Hin + iy) * Win + ix) * Cin + ci];
      }
      lA[buf][r][kkA] = us;
    }
    const int cc = n0 + cB;
#pragma unroll
    for (int it = 0; it < 16; ++it) {
      int kr = krB0 + it * 2;
      unsigned short us = 0;
      if (cc < Cout) us = Bp[(long long)(k0 + kr) * Cout + cc];
      lB[buf][cB][kr] = us;
    }
  };

  loadTiles(0, 0);
  __syncthreads();
  for (int s = 0; s < NSTEP; ++s) {
    const int cur = s & 1;
    if (s + 1 < NSTEP) loadTiles((s + 1) * 32, cur ^ 1);
    wmma_step(lA[cur], lB[cur], wv, grp, lr, acc);
    __syncthreads();
  }

  constexpr int OH = 2 * Hin, OW = 2 * Win;
#pragma unroll
  for (int j = 0; j < 4; ++j) {
    int c = n0 + j * 16 + lr;
    if (c >= Cout) continue;
    float bb = bias[c];
#pragma unroll
    for (int v = 0; v < 8; ++v) {
      int m = m0 + wv * 16 + grp * 8 + v;
      if (m >= Mtot) continue;
      int x = m % Win; int tm = m / Win;
      int y = tm % Hin; int n = tm / Hin;
      int oh = 2 * y + py, ow = 2 * x + px;
      float val = acc[j][v] + bb;
      if (tanh_flag) val = tanhf(val);
      if (OutNCHW)
        OutNCHW[(((long long)n * Cout + c) * OH + oh) * OW + ow] = val;
      else
        OutNHWC[(((long long)n * OH + oh) * OW + ow) * Cout + c] = val;
    }
  }
}

// ---------------------------------------------------------------------------
// VQ distance GEMM: dots[m][n] = ||e_n||^2 - 2 * <z_m, e_n>
// A-tile staged via the Tensor Data Mover (TDM): dense 64x32 bf16 tile,
// per-row LDS padding 64B data + 4B pad == lA row stride (68B). Wave 0 issues
// the DMA; completion via s_wait_tensorcnt before the tile barrier.
// ---------------------------------------------------------------------------
template <int M, int Nn, int K>
__global__ __launch_bounds__(128)
void k_gemm_vq(const unsigned short* __restrict__ A,   // [M][K] bf16
               const unsigned short* __restrict__ B,   // [K][Nn] bf16
               const float* __restrict__ esq,
               float* __restrict__ dots) {
  constexpr int NSTEP = K / 32;
  __shared__ unsigned short lA[2][64][34];
  __shared__ unsigned short lB[2][64][34];
  const int tid = threadIdx.x;
  const int wv = tid >> 5, lane = tid & 31;
  const int grp = lane >> 4, lr = lane & 15;
  const int m0 = blockIdx.x * 64;
  const int n0 = blockIdx.y * 64;
  const int kkA = tid & 31;
  const int rA0 = tid >> 5;
  const int cB  = tid & 63;
  const int krB0 = tid >> 6;

  v8f acc[4];
#pragma unroll
  for (int j = 0; j < 4; ++j)
#pragma unroll
    for (int e = 0; e < 8; ++e) acc[j][e] = 0.f;

#if USE_TDM
  // TDM descriptor: 2D tensor, data_size=2B, tile 32(k) x 64(m),
  // tensor_dim0_stride = K, LDS pad: 1 dword after every 16 dwords (one row).
  auto loadA_tdm = [&](int k0, int buf) {
    unsigned long long ga =
        (unsigned long long)(const void*)(A + (long long)m0 * K + k0);
    unsigned lds_off = (unsigned)(unsigned long long)(const void*)&lA[buf][0][0];
    v4u g0;
    g0[0] = 1u;                                    // count=1 (valid user D#)
    g0[1] = lds_off;                               // lds_addr
    g0[2] = (unsigned)ga;                          // global_addr[31:0]
    g0[3] = (unsigned)((ga >> 32) & 0x1FFFFFFu) | (2u << 30);  // addr[56:32]|type=2
    v8i g1;
    g1[0] = (int)((1u << 16) | (1u << 20) | (3u << 22));  // 2B, pad_en, 16dw, 1dw
    g1[1] = (int)(32u << 16);                      // tensor_dim0 = 32
    g1[2] = (int)(64u << 16);                      // tensor_dim1 = 64
    g1[3] = (int)(32u << 16);                      // tile_dim0 = 32
    g1[4] = (int)64u;                              // tile_dim1 = 64, tile_dim2 = 0
    g1[5] = (int)K;                                // tensor_dim0_stride = K
    g1[6] = 0;
    g1[7] = 0;
    v4i gz = {0, 0, 0, 0};
#if defined(__clang_major__) && __clang_major__ >= 23
    v8i gz8 = {0, 0, 0, 0, 0, 0, 0, 0};
    __builtin_amdgcn_tensor_load_to_lds(g0, g1, gz, gz, gz8, 0);
#else
    __builtin_amdgcn_tensor_load_to_lds(g0, g1, gz, gz, 0);
#endif
  };
#else
  auto loadA_man = [&](int k0, int buf) {
#pragma unroll
    for (int it = 0; it < 16; ++it) {
      int r = rA0 + it * 4;
      int m = m0 + r;
      lA[buf][r][kkA] = (m < M) ? A[(long long)m * K + k0 + kkA] : (unsigned short)0;
    }
  };
#endif
  auto loadB = [&](int k0, int buf) {
    const int cc = n0 + cB;
#pragma unroll
    for (int it = 0; it < 16; ++it) {
      int kr = krB0 + it * 2;
      lB[buf][cB][kr] = (cc < Nn) ? B[(long long)(k0 + kr) * Nn + cc] : (unsigned short)0;
    }
  };

#if USE_TDM
  if (wv == 0) loadA_tdm(0, 0);
  loadB(0, 0);
  if (wv == 0) __builtin_amdgcn_s_wait_tensorcnt(0);
  __syncthreads();
  for (int s = 0; s < NSTEP; ++s) {
    const int cur = s & 1;
    if (s + 1 < NSTEP) {
      if (wv == 0) loadA_tdm((s + 1) * 32, cur ^ 1);
      loadB((s + 1) * 32, cur ^ 1);
    }
    wmma_step(lA[cur], lB[cur], wv, grp, lr, acc);
    if (wv == 0 && s + 1 < NSTEP) __builtin_amdgcn_s_wait_tensorcnt(0);
    __syncthreads();
  }
#else
  loadA_man(0, 0);
  loadB(0, 0);
  __syncthreads();
  for (int s = 0; s < NSTEP; ++s) {
    const int cur = s & 1;
    if (s + 1 < NSTEP) { loadA_man((s + 1) * 32, cur ^ 1); loadB((s + 1) * 32, cur ^ 1); }
    wmma_step(lA[cur], lB[cur], wv, grp, lr, acc);
    __syncthreads();
  }
#endif

#pragma unroll
  for (int j = 0; j < 4; ++j) {
    int c = n0 + j * 16 + lr;
    if (c >= Nn) continue;
#pragma unroll
    for (int v = 0; v < 8; ++v) {
      int m = m0 + wv * 16 + grp * 8 + v;
      if (m >= M) continue;
      dots[(long long)m * Nn + c] = esq[c] - 2.f * acc[j][v];
    }
  }
}

// ---------------------------------------------------------------------------
// VQ codebook prep + argmin/gather
// ---------------------------------------------------------------------------
__global__ void k_emb_prep(const float* __restrict__ emb,
                           unsigned short* __restrict__ embB,  // [C=256][K=512]
                           float* __restrict__ esq, int C, int Kn) {
  int i = blockIdx.x * blockDim.x + threadIdx.x;
  if (i < C * Kn) {
    int n = i % Kn, c = i / Kn;
    embB[i] = f2us_bf16(emb[(long long)n * C + c]);
  }
  if (i < Kn) {
    float s = 0.f;
    for (int c = 0; c < C; ++c) { float v = emb[(long long)i * C + c]; s += v * v; }
    esq[i] = s;
  }
}

__global__ __launch_bounds__(256)
void k_argmin_gather(const float* __restrict__ dots,
                     const float* __restrict__ emb,
                     float* __restrict__ zq_f32,
                     unsigned short* __restrict__ zq_bf,
                     int* __restrict__ latents, int Nn, int C) {
  __shared__ float sv[256];
  __shared__ int   si[256];
  const int m = blockIdx.x, t = threadIdx.x;
  float best = 3.4e38f; int bi = 0;
  for (int n = t; n < Nn; n += 256) {
    float v = dots[(long long)m * Nn + n];
    if (v < best) { best = v; bi = n; }
  }
  sv[t] = best; si[t] = bi;
  __syncthreads();
  for (int s = 128; s > 0; s >>= 1) {
    if (t < s) {
      if (sv[t + s] < sv[t] || (sv[t + s] == sv[t] && si[t + s] < si[t])) {
        sv[t] = sv[t + s]; si[t] = si[t + s];
      }
    }
    __syncthreads();
  }
  const int lat = si[0];
  if (t == 0) latents[m] = lat;
  for (int c = t; c < C; c += 256) {
    float v = emb[(long long)lat * C + c];
    zq_f32[(long long)m * C + c] = v;
    if (zq_bf) zq_bf[(long long)m * C + c] = f2us_bf16(v);
  }
}

// ---------------------------------------------------------------------------
// Host-side templated launchers
// ---------------------------------------------------------------------------
template <int N, int H, int W, int Cin, int Cout, int KH, int KW, int ST, int PD>
static inline void launch_conv(hipStream_t stream, unsigned short* Wg,
                               const unsigned short* X, const float* wsrc,
                               const float* bias, const float* resid,
                               float* out, int relu) {
  constexpr int OH = (H + 2 * PD - KH) / ST + 1;
  constexpr int OW = (W + 2 * PD - KW) / ST + 1;
  constexpr int Ktot = Cin * KH * KW;
  constexpr int Kpad = (Ktot + 31) & ~31;
  constexpr int M = N * OH * OW;
  k_prep_wconv<<<dim3((Kpad * Cout + 255) / 256), dim3(256), 0, stream>>>(
      wsrc, Wg, Cout, Cin, KH, KW, Kpad);
  k_conv_wmma<N, H, W, Cin, Cout, KH, KW, ST, PD>
      <<<dim3((M + 63) / 64, (Cout + 63) / 64, 1), dim3(128), 0, stream>>>(
          X, Wg, bias, resid, out, relu);
}

template <int N, int Hin, int Win, int Cin, int Cout>
static inline void launch_deconv(hipStream_t stream, unsigned short* Wg,
                                 const unsigned short* X, const float* wsrc,
                                 const float* bias, float* outNHWC,
                                 float* outNCHW, int tanh_flag) {
  constexpr int M = N * Hin * Win;
  k_prep_wdeconv<<<dim3((4 * Cin * Cout + 255) / 256, 4), dim3(256), 0, stream>>>(
      wsrc, Wg, Cin, Cout);
  k_deconv_wmma<N, Hin, Win, Cin, Cout>
      <<<dim3((M + 63) / 64, (Cout + 63) / 64, 4), dim3(128), 0, stream>>>(
          X, Wg, bias, outNHWC, outNCHW, tanh_flag);
}

// ---------------------------------------------------------------------------
// Host orchestration
// ---------------------------------------------------------------------------
extern "C" void kernel_launch(void* const* d_in, const int* in_sizes, int n_in,
                              void* d_out, int out_size, void* d_ws, size_t ws_size,
                              hipStream_t stream) {
  (void)in_sizes; (void)n_in; (void)out_size; (void)ws_size;
  int a = 0;
  const float* x      = (const float*)d_in[a++];   // (16,3,128,128)
  const float* c1_w   = (const float*)d_in[a++];
  const float* c1_b   = (const float*)d_in[a++];
  const float* bn1_g  = (const float*)d_in[a++];
  const float* bn1_b  = (const float*)d_in[a++];
  const float* c2_w   = (const float*)d_in[a++];
  const float* c2_b   = (const float*)d_in[a++];
  const float* bn2_g  = (const float*)d_in[a++];
  const float* bn2_b  = (const float*)d_in[a++];
  const float* r1_w1  = (const float*)d_in[a++];
  const float* r1_b1  = (const float*)d_in[a++];
  const float* r1_g   = (const float*)d_in[a++];
  const float* r1_b   = (const float*)d_in[a++];
  const float* r1_w2  = (const float*)d_in[a++];
  const float* r1_b2  = (const float*)d_in[a++];
  const float* bn3_g  = (const float*)d_in[a++];
  const float* bn3_b  = (const float*)d_in[a++];
  const float* r2_w1  = (const float*)d_in[a++];
  const float* r2_b1  = (const float*)d_in[a++];
  const float* r2_g   = (const float*)d_in[a++];
  const float* r2_b   = (const float*)d_in[a++];
  const float* r2_w2  = (const float*)d_in[a++];
  const float* r2_b2  = (const float*)d_in[a++];
  const float* bn4_g  = (const float*)d_in[a++];
  const float* bn4_b  = (const float*)d_in[a++];
  const float* emb    = (const float*)d_in[a++];   // (512,256)
  const float* dr1_w1 = (const float*)d_in[a++];
  const float* dr1_b1 = (const float*)d_in[a++];
  const float* dr1_g  = (const float*)d_in[a++];
  const float* dr1_b  = (const float*)d_in[a++];
  const float* dr1_w2 = (const float*)d_in[a++];
  const float* dr1_b2 = (const float*)d_in[a++];
  const float* dbn1_g = (const float*)d_in[a++];
  const float* dbn1_b = (const float*)d_in[a++];
  const float* dr2_w1 = (const float*)d_in[a++];
  const float* dr2_b1 = (const float*)d_in[a++];
  const float* dr2_g  = (const float*)d_in[a++];
  const float* dr2_b  = (const float*)d_in[a++];
  const float* dr2_w2 = (const float*)d_in[a++];
  const float* dr2_b2 = (const float*)d_in[a++];
  const float* dbn2_g = (const float*)d_in[a++];
  const float* dbn2_b = (const float*)d_in[a++];
  const float* d1_w   = (const float*)d_in[a++];   // (256,256,4,4) Cin,Cout
  const float* d1_b   = (const float*)d_in[a++];
  const float* dbn3_g = (const float*)d_in[a++];
  const float* dbn3_b = (const float*)d_in[a++];
  const float* d2_w   = (const float*)d_in[a++];   // (256,3,4,4)
  const float* d2_b   = (const float*)d_in[a++];

  float* out_x  = (float*)d_out;                 // x_tilde 16*3*128*128
  float* out_ze = out_x + 786432ll;              // z_e_x   16*256*32*32
  float* out_zq = out_ze + 4194304ll;            // z_q_x

  // ------ workspace carve-out ------
  char* ws = (char*)d_ws;
  size_t off = 0;
  auto carve = [&](size_t bytes) -> char* {
    char* p = ws + off;
    off += (bytes + 255) & ~(size_t)255;
    return p;
  };
  const long long S64 = 16ll * 64 * 64 * 256;    // 16.7M elems
  const long long S32 = 16ll * 32 * 32 * 256;    // 4.19M elems
  float*          F64a = (float*)carve(S64 * 4);
  unsigned short* BF64 = (unsigned short*)carve(S64 * 2);
  float*          P    = (float*)carve(S32 * 4);
  float*          Q    = (float*)carve(S32 * 4);
  float*          R    = (float*)carve(S32 * 4);
  unsigned short* T16  = (unsigned short*)carve(S32 * 2);
  unsigned short* Wg   = (unsigned short*)carve(4ll * 1024 * 256 * 2);
  float*          stats = (float*)carve(4096);
  unsigned short* embB = (unsigned short*)carve(256ll * 512 * 2);
  float*          esq  = (float*)carve(512 * 4);
  int*            lat  = (int*)carve(16384 * 4);
  unsigned short* inbf = (unsigned short*)carve(786432ll * 2);
  float*          dots = F64a;   // overlap: free between conv2 and deconv d1

  auto bn = [&](const float* xin, const float* g_, const float* b_,
                float* of32, unsigned short* obf, int M, int relu) {
    hipMemsetAsync(stats, 0, 2 * 256 * sizeof(float), stream);
    k_bn_stats<<<dim3(256), dim3(256), 0, stream>>>(xin, stats, M, 256);
    long long tot = (long long)M * 256;
    k_bn_apply<<<dim3((unsigned)((tot + 255) / 256)), dim3(256), 0, stream>>>(
        xin, stats, g_, b_, of32, obf, tot, 256, 1.f / (float)M, relu);
  };
  auto cvt = [&](const float* in, float* of32, unsigned short* obf,
                 long long tot, int relu) {
    k_convert<<<dim3((unsigned)((tot + 255) / 256)), dim3(256), 0, stream>>>(
        in, of32, obf, tot, relu);
  };

  // ================= encoder =================
  k_nchw_to_nhwc_bf16<<<dim3((786432 + 255) / 256), dim3(256), 0, stream>>>(
      x, inbf, 16, 3, 128, 128);
  launch_conv<16, 128, 128, 3, 256, 4, 4, 2, 1>(stream, Wg, inbf, c1_w, c1_b, nullptr, F64a, 0);
  bn(F64a, bn1_g, bn1_b, nullptr, BF64, 16 * 64 * 64, 1);                 // relu
  launch_conv<16, 64, 64, 256, 256, 4, 4, 2, 1>(stream, Wg, BF64, c2_w, c2_b, nullptr, P, 0);
  bn(P, bn2_g, bn2_b, P, nullptr, 16 * 32 * 32, 0);
  // resblock r1: P -> R
  cvt(P, nullptr, T16, S32, 1);
  launch_conv<16, 32, 32, 256, 256, 3, 3, 1, 1>(stream, Wg, T16, r1_w1, r1_b1, nullptr, Q, 0);
  bn(Q, r1_g, r1_b, nullptr, T16, 16 * 32 * 32, 1);
  launch_conv<16, 32, 32, 256, 256, 1, 1, 1, 0>(stream, Wg, T16, r1_w2, r1_b2, P, R, 0);
  bn(R, bn3_g, bn3_b, R, nullptr, 16 * 32 * 32, 0);
  // resblock r2: R -> P
  cvt(R, nullptr, T16, S32, 1);
  launch_conv<16, 32, 32, 256, 256, 3, 3, 1, 1>(stream, Wg, T16, r2_w1, r2_b1, nullptr, Q, 0);
  bn(Q, r2_g, r2_b, nullptr, T16, 16 * 32 * 32, 1);
  launch_conv<16, 32, 32, 256, 256, 1, 1, 1, 0>(stream, Wg, T16, r2_w2, r2_b2, R, P, 0);
  bn(P, bn4_g, bn4_b, P, T16, 16 * 32 * 32, 0);          // z_e_x fp32 + bf16 zf
  k_nhwc_to_nchw<<<dim3((unsigned)((S32 + 255) / 256)), dim3(256), 0, stream>>>(
      P, out_ze, 16, 256, 32, 32);

  // ================= vector quantization =================
  k_emb_prep<<<dim3((256 * 512 + 255) / 256), dim3(256), 0, stream>>>(
      emb, embB, esq, 256, 512);
  k_gemm_vq<16384, 512, 256><<<dim3(16384 / 64, 512 / 64, 1), dim3(128), 0, stream>>>(
      T16, embB, esq, dots);
  k_argmin_gather<<<dim3(16384), dim3(256), 0, stream>>>(
      dots, emb, Q, nullptr, lat, 512, 256);             // z_q fp32 -> Q
  k_nhwc_to_nchw<<<dim3((unsigned)((S32 + 255) / 256)), dim3(256), 0, stream>>>(
      Q, out_zq, 16, 256, 32, 32);

  // ================= decoder =================
  // resblock dr1: Q -> P
  cvt(Q, nullptr, T16, S32, 1);
  launch_conv<16, 32, 32, 256, 256, 3, 3, 1, 1>(stream, Wg, T16, dr1_w1, dr1_b1, nullptr, R, 0);
  bn(R, dr1_g, dr1_b, nullptr, T16, 16 * 32 * 32, 1);
  launch_conv<16, 32, 32, 256, 256, 1, 1, 1, 0>(stream, Wg, T16, dr1_w2, dr1_b2, Q, P, 0);
  bn(P, dbn1_g, dbn1_b, P, nullptr, 16 * 32 * 32, 0);
  // resblock dr2: P -> R
  cvt(P, nullptr, T16, S32, 1);
  launch_conv<16, 32, 32, 256, 256, 3, 3, 1, 1>(stream, Wg, T16, dr2_w1, dr2_b1, nullptr, Q, 0);
  bn(Q, dr2_g, dr2_b, nullptr, T16, 16 * 32 * 32, 1);
  launch_conv<16, 32, 32, 256, 256, 1, 1, 1, 0>(stream, Wg, T16, dr2_w2, dr2_b2, P, R, 0);
  bn(R, dbn2_g, dbn2_b, nullptr, T16, 16 * 32 * 32, 1);  // bn2 + relu -> bf16
  // deconv d1: 32x32x256 -> 64x64x256 (NHWC fp32)
  launch_deconv<16, 32, 32, 256, 256>(stream, Wg, T16, d1_w, d1_b, F64a, nullptr, 0);
  bn(F64a, dbn3_g, dbn3_b, nullptr, BF64, 16 * 64 * 64, 1);  // bn3 + relu
  // deconv d2: 64x64x256 -> 128x128x3, tanh, NCHW straight into d_out
  launch_deconv<16, 64, 64, 256, 3>(stream, Wg, BF64, d2_w, d2_b, nullptr, out_x, 1);
}
